// WindowAttention_30245159699019
// MI455X (gfx1250) — matrix operand order repaired and checked by reading, not compiled, not run on hardware
//
#include <hip/hip_runtime.h>

typedef unsigned int u32;
typedef unsigned short u16;
typedef unsigned long long u64;
typedef __bf16 bf16_t;
typedef bf16_t bf16x16 __attribute__((ext_vector_type(16)));
typedef u32 u32x8 __attribute__((ext_vector_type(8)));
typedef float f32x8 __attribute__((ext_vector_type(8)));

#define DEV __device__ __forceinline__

// ---------- bf16 helpers (round-to-nearest-even) ----------
DEV u16 f2bf(float f) {
    u32 u = __float_as_uint(f);
    u += 0x7FFFu + ((u >> 16) & 1u);
    return (u16)(u >> 16);
}

// ---------- WMMA wrapper: D = A(16x32 bf16) * B(32x16 bf16) + C(f32) ----------
DEV f32x8 wmma_bf16(u32x8 a, u32x8 b, f32x8 c) {
    return __builtin_amdgcn_wmma_f32_16x16x32_bf16(
        false, __builtin_bit_cast(bf16x16, a),
        false, __builtin_bit_cast(bf16x16, b),
        (short)0, c, false, false);
}

// A fragment (16x32 bf16). LDS row-major [m][k]; row_stride32 = K-stride in u32 words.
// ISA layout: lanes 0-15 row M=lane hold K {0..7, 16..23}; lanes 16-31 hold K {8..15, 24..31}.
DEV u32x8 load_a_frag(const u32* lds, int row_stride32, int m0, int lane) {
    const u32* row = lds + (m0 + (lane & 15)) * row_stride32 + ((lane < 16) ? 0 : 4);
    u32x8 f;
    f[0] = row[0]; f[1] = row[1]; f[2] = row[2]; f[3] = row[3];
    f[4] = row[8]; f[5] = row[9]; f[6] = row[10]; f[7] = row[11];
    return f;
}

// B fragment (32x16 bf16, KxN). LDS stored transposed [n][k] so K-pairs are contiguous.
// ISA layout: lanes 0-15 col N=lane hold K {0..7, 16..23}; lanes 16-31 hold K {8..15, 24..31}.
DEV u32x8 load_b_frag(const u32* lds, int row_stride32, int n0, int lane) {
    const u32* row = lds + (n0 + (lane & 15)) * row_stride32 + ((lane < 16) ? 0 : 4);
    u32x8 f;
    f[0] = row[0]; f[1] = row[1]; f[2] = row[2]; f[3] = row[3];
    f[4] = row[8]; f[5] = row[9]; f[6] = row[10]; f[7] = row[11];
    return f;
}

// LDS byte address of a generic pointer to __shared__ (flat LDS aperture: low 32 bits = LDS offset)
DEV u32 lds_addr_of(const void* p) { return (u32)(size_t)p; }

// ---------- kernel 1: f32 -> bf16 convert (vectorized) ----------
__global__ void cvt_f32_bf16(const float4* __restrict__ src, u64* __restrict__ dst, int n4) {
    int i = blockIdx.x * blockDim.x + threadIdx.x;
    int stride = gridDim.x * blockDim.x;
    for (; i < n4; i += stride) {
        float4 f = src[i];
        u64 p = (u64)f2bf(f.x) | ((u64)f2bf(f.y) << 16) |
                ((u64)f2bf(f.z) << 32) | ((u64)f2bf(f.w) << 48);
        dst[i] = p;
    }
}

// ---------- shared GEMM core: C[128x128] tile, K=512, BK=32, 8 waves ----------
// Double-buffered LDS. A tile (contiguous rows) streams via GLOBAL_LOAD_ASYNC_TO_LDS_B128
// (ASYNCcnt); B tile goes through VGPRs to be transposed to [n][k] for B-fragment layout.
// As/Bs: 2 buffers x 128x32 bf16 each (8KB per buffer).
DEV void gemm_core(const u16* __restrict__ A, const u16* __restrict__ Bm, int nstride,
                   int bx, int by, int t, u16* As, u16* Bs, f32x8 (&acc)[4][2]) {
    const int lane = t & 31, w = t >> 5;
    const int wm = w & 1, wn = w >> 1;

    // --- per-thread A async-copy addresses (2 x 16B chunks per thread per step) ---
    const int cA0 = t & 3, rA0 = t >> 2;                // chunk t         (rows 0..63)
    const int cA1 = t & 3, rA1 = (t + 256) >> 2;        // chunk t+256     (rows 64..127)
    const u32 ldsA = lds_addr_of(As);
    const u32 lA0 = ldsA + (u32)t * 16;
    const u32 lA1 = ldsA + (u32)(t + 256) * 16;
    const u32 gA0 = (u32)(((by * 128 + rA0) * 512 + cA0 * 8) * 2);   // byte offsets
    const u32 gA1 = (u32)(((by * 128 + rA1) * 512 + cA1 * 8) * 2);

    // --- per-thread B chunk coordinates ---
    const int rB0 = t >> 4, cB = t & 15;                // k-rows 0..15
    const int rB1 = rB0 + 16;                           // k-rows 16..31

    uint4 bregs[2];

    auto issue_asyncA = [&](int buf, int kk) {
        u32 l0 = lA0 + (u32)buf * 8192u;
        u32 l1 = lA1 + (u32)buf * 8192u;
        u32 g0 = gA0 + (u32)kk * 64u;
        u32 g1 = gA1 + (u32)kk * 64u;
        asm volatile("global_load_async_to_lds_b128 %0, %1, %4\n\t"
                     "global_load_async_to_lds_b128 %2, %3, %4"
                     :: "v"(l0), "v"(g0), "v"(l1), "v"(g1), "s"(A)
                     : "memory");
    };
    auto load_Bregs = [&](int kk) {
        const u16* base = Bm + (size_t)(kk * 32) * nstride + bx * 128 + cB * 8;
        bregs[0] = *(const uint4*)(base + (size_t)rB0 * nstride);
        bregs[1] = *(const uint4*)(base + (size_t)rB1 * nstride);
    };
    auto store_Bregs = [&](int buf) {
        u16* dst = Bs + buf * (128 * 32);
        u16 e[8];
        *(uint4*)e = bregs[0];
#pragma unroll
        for (int j = 0; j < 8; ++j) dst[(cB * 8 + j) * 32 + rB0] = e[j];
        *(uint4*)e = bregs[1];
#pragma unroll
        for (int j = 0; j < 8; ++j) dst[(cB * 8 + j) * 32 + rB1] = e[j];
    };

    // prologue: fill buffer 0 for step 0
    issue_asyncA(0, 0);
    load_Bregs(0);
    store_Bregs(0);
    asm volatile("s_wait_asynccnt 0x0" ::: "memory");
    __syncthreads();

    for (int kk = 0; kk < 16; ++kk) {
        const int buf = kk & 1, nbuf = buf ^ 1;
        if (kk < 15) {                       // prefetch step kk+1 into the other buffer
            issue_asyncA(nbuf, kk + 1);
            load_Bregs(kk + 1);
            store_Bregs(nbuf);
        }
        const u32* As32 = (const u32*)(As + buf * (128 * 32));
        const u32* Bs32 = (const u32*)(Bs + buf * (128 * 32));
        u32x8 af[4];
#pragma unroll
        for (int mt = 0; mt < 4; ++mt) af[mt] = load_a_frag(As32, 16, wm * 64 + mt * 16, lane);
#pragma unroll
        for (int nt = 0; nt < 2; ++nt) {
            u32x8 bfr = load_b_frag(Bs32, 16, wn * 32 + nt * 16, lane);
#pragma unroll
            for (int mt = 0; mt < 4; ++mt) acc[mt][nt] = wmma_bf16(af[mt], bfr, acc[mt][nt]);
        }
        if (kk < 15) {
            asm volatile("s_wait_asynccnt 0x0" ::: "memory");
            __syncthreads();                 // also waits dscnt for B stores & frag loads
        }
    }
}

// ---------- kernel 2: QKV GEMM + bias + q-scale, scatter to [b,h,n,d] bf16 ----------
__global__ __launch_bounds__(256) void gemm_qkv_kernel(
    const u16* __restrict__ A, const u16* __restrict__ Bm, const float* __restrict__ bias,
    u16* __restrict__ qb, u16* __restrict__ kb, u16* __restrict__ vb) {
    __shared__ u16 As[2 * 128 * 32];
    __shared__ u16 Bs[2 * 128 * 32];
    const int bx = blockIdx.x, by = blockIdx.y, t = threadIdx.x;
    const int lane = t & 31, w = t >> 5, wm = w & 1, wn = w >> 1;
    f32x8 acc[4][2] = {};
    gemm_core(A, Bm, 1536, bx, by, t, As, Bs, acc);
    const float scale = 0.17677669529663687f;  // 32^-0.5
    const int rbase = 8 * (lane >> 4), cbase = lane & 15;
#pragma unroll
    for (int mt = 0; mt < 4; ++mt)
#pragma unroll
        for (int nt = 0; nt < 2; ++nt)
#pragma unroll
            for (int r = 0; r < 8; ++r) {
                int grow = by * 128 + wm * 64 + mt * 16 + rbase + r;
                int gcol = bx * 128 + wn * 32 + nt * 16 + cbase;
                float v = acc[mt][nt][r] + bias[gcol];
                int s = gcol >> 9;           // 0=q 1=k 2=v
                int rem = gcol & 511;
                int h = rem >> 5, d = rem & 31;
                int b = grow / 49, ntok = grow - b * 49;
                if (s == 0) v *= scale;
                u16* dst = (s == 0) ? qb : ((s == 1) ? kb : vb);
                dst[(((size_t)b * 16 + h) * 49 + ntok) * 32 + d] = f2bf(v);
            }
}

// ---------- kernel 3: per-(window,head) attention ----------
// block = 128 threads (4 waves); wave w owns score rows 16w..16w+15.
__global__ __launch_bounds__(128) void attn_kernel(
    const u16* __restrict__ qb, const u16* __restrict__ kb, const u16* __restrict__ vb,
    const float* __restrict__ rpb, const int* __restrict__ relidx,
    const float* __restrict__ mask, u16* __restrict__ out) {
    __shared__ u16 qs[64 * 32];   // [tok][d]
    __shared__ u16 ks[64 * 32];   // [tok][d]  (== B^T layout for S = Q K^T)
    __shared__ u16 vT[32 * 64];   // [d][tok]  (== B^T layout for O = P V)
    __shared__ u16 ps[64 * 64];   // probabilities, [m][j]
    const int bh = blockIdx.x;
    const int b = bh >> 4, h = bh & 15;
    const int win = b & 63;       // b % NUM_WIN
    const int t = threadIdx.x, lane = t & 31, w = t >> 5;
    const u32* gq = (const u32*)qb + (size_t)bh * 784;  // 49*32/2 u32 words
    const u32* gk = (const u32*)kb + (size_t)bh * 784;
    const u16* gv = vb + (size_t)bh * 49 * 32;
    u32* qs32 = (u32*)qs;
    u32* ks32 = (u32*)ks;
#pragma unroll
    for (int i = 0; i < 8; ++i) {
        int e = t + i * 128;            // 1024 u32 words
        int tok = e >> 4;
        qs32[e] = (tok < 49) ? gq[e] : 0u;
        ks32[e] = (tok < 49) ? gk[e] : 0u;
    }
#pragma unroll
    for (int i = 0; i < 16; ++i) {
        int e = t + i * 128;            // 2048 u16, transpose on the fly
        int tok = e >> 5, d = e & 31;
        vT[d * 64 + tok] = (tok < 49) ? gv[e] : (u16)0;
    }
    __syncthreads();

    // S strip (16x64): K=32 single WMMA step, 4 N-tiles
    f32x8 s[4] = {};
    {
        u32x8 aq = load_a_frag((const u32*)qs, 16, w * 16, lane);
#pragma unroll
        for (int nt = 0; nt < 4; ++nt) {
            u32x8 bk = load_b_frag((const u32*)ks, 16, nt * 16, lane);
            s[nt] = wmma_bf16(aq, bk, s[nt]);
        }
    }
    const int rbase = 8 * (lane >> 4), jn = lane & 15;
    // bias + mask + padding
#pragma unroll
    for (int nt = 0; nt < 4; ++nt) {
        int j = nt * 16 + jn;
#pragma unroll
        for (int r = 0; r < 8; ++r) {
            int m = w * 16 + rbase + r;
            if (j < 49) {
                if (m < 49) {
                    int ridx = relidx[m * 49 + j];
                    s[nt][r] += rpb[ridx * 16 + h] + mask[((size_t)win * 49 + m) * 49 + j];
                }
            } else {
                s[nt][r] = -__builtin_inff();
            }
        }
    }
    // row softmax: each row's 64 cols live on 16 lanes (per half-wave) x 4 tiles
#pragma unroll
    for (int r = 0; r < 8; ++r) {
        float mx = fmaxf(fmaxf(s[0][r], s[1][r]), fmaxf(s[2][r], s[3][r]));
#pragma unroll
        for (int off = 1; off < 16; off <<= 1) mx = fmaxf(mx, __shfl_xor(mx, off, 16));
        float e0 = __expf(s[0][r] - mx), e1 = __expf(s[1][r] - mx);
        float e2 = __expf(s[2][r] - mx), e3 = __expf(s[3][r] - mx);
        float sum = (e0 + e1) + (e2 + e3);
#pragma unroll
        for (int off = 1; off < 16; off <<= 1) sum += __shfl_xor(sum, off, 16);
        float inv = 1.0f / sum;
        s[0][r] = e0 * inv; s[1][r] = e1 * inv; s[2][r] = e2 * inv; s[3][r] = e3 * inv;
    }
    // spill P to LDS to re-layout C-tile -> A-fragment
#pragma unroll
    for (int nt = 0; nt < 4; ++nt) {
        int j = nt * 16 + jn;
#pragma unroll
        for (int r = 0; r < 8; ++r) ps[(w * 16 + rbase + r) * 64 + j] = f2bf(s[nt][r]);
    }
    __syncthreads();
    // O = P * V : 2 K-steps of 32, 2 N-tiles of 16
    f32x8 o[2] = {};
#pragma unroll
    for (int kstep = 0; kstep < 2; ++kstep) {
        u32x8 ap = load_a_frag((const u32*)ps + kstep * 16, 32, w * 16, lane);
#pragma unroll
        for (int nt = 0; nt < 2; ++nt) {
            u32x8 bv = load_b_frag((const u32*)vT + kstep * 16, 32, nt * 16, lane);
            o[nt] = wmma_bf16(ap, bv, o[nt]);
        }
    }
    // store to attn_out [b*49+m][h*32+d] bf16 (row-major M x 512)
#pragma unroll
    for (int nt = 0; nt < 2; ++nt) {
        int d = nt * 16 + jn;
#pragma unroll
        for (int r = 0; r < 8; ++r) {
            int m = w * 16 + rbase + r;
            if (m < 49)
                out[((size_t)b * 49 + m) * 512 + h * 32 + d] = f2bf(o[nt][r]);
        }
    }
}

// ---------- kernel 4: output projection GEMM, f32 result + bias ----------
__global__ __launch_bounds__(256) void gemm_proj_kernel(
    const u16* __restrict__ A, const u16* __restrict__ Bm, const float* __restrict__ bias,
    float* __restrict__ out) {
    __shared__ u16 As[2 * 128 * 32];
    __shared__ u16 Bs[2 * 128 * 32];
    const int bx = blockIdx.x, by = blockIdx.y, t = threadIdx.x;
    const int lane = t & 31, w = t >> 5, wm = w & 1, wn = w >> 1;
    f32x8 acc[4][2] = {};
    gemm_core(A, Bm, 512, bx, by, t, As, Bs, acc);
    const int rbase = 8 * (lane >> 4), cbase = lane & 15;
#pragma unroll
    for (int mt = 0; mt < 4; ++mt)
#pragma unroll
        for (int nt = 0; nt < 2; ++nt)
#pragma unroll
            for (int r = 0; r < 8; ++r) {
                int grow = by * 128 + wm * 64 + mt * 16 + rbase + r;
                int gcol = bx * 128 + wn * 32 + nt * 16 + cbase;
                out[(size_t)grow * 512 + gcol] = acc[mt][nt][r] + bias[gcol];
            }
}

extern "C" void kernel_launch(void* const* d_in, const int* in_sizes, int n_in,
                              void* d_out, int out_size, void* d_ws, size_t ws_size,
                              hipStream_t stream) {
    (void)in_sizes; (void)n_in; (void)out_size; (void)ws_size;
    const float* x      = (const float*)d_in[0];
    const float* mask   = (const float*)d_in[1];
    const float* w_qkv  = (const float*)d_in[2];
    const float* b_qkv  = (const float*)d_in[3];
    const float* w_proj = (const float*)d_in[4];
    const float* b_proj = (const float*)d_in[5];
    const float* rpb    = (const float*)d_in[6];
    const int*   relidx = (const int*)d_in[7];

    char* ws = (char*)d_ws;
    size_t off = 0;
    const size_t NX = (size_t)100352 * 512;  // 2048*49*512 elements
    u16* x_bf    = (u16*)(ws + off); off += NX * 2;
    u16* wqkv_bf = (u16*)(ws + off); off += (size_t)512 * 1536 * 2;
    u16* wproj_bf= (u16*)(ws + off); off += (size_t)512 * 512 * 2;
    u16* qbuf    = (u16*)(ws + off); off += NX * 2;   // [b,h,49,32] == NX elements
    u16* kbuf    = (u16*)(ws + off); off += NX * 2;
    u16* vbuf    = (u16*)(ws + off); off += NX * 2;
    u16* attn    = (u16*)(ws + off); off += NX * 2;

    cvt_f32_bf16<<<2048, 256, 0, stream>>>((const float4*)x, (u64*)x_bf, (int)(NX / 4));
    cvt_f32_bf16<<<192, 256, 0, stream>>>((const float4*)w_qkv, (u64*)wqkv_bf, 512 * 1536 / 4);
    cvt_f32_bf16<<<64, 256, 0, stream>>>((const float4*)w_proj, (u64*)wproj_bf, 512 * 512 / 4);

    gemm_qkv_kernel<<<dim3(12, 784), 256, 0, stream>>>(x_bf, wqkv_bf, b_qkv, qbuf, kbuf, vbuf);
    attn_kernel<<<32768, 128, 0, stream>>>(qbuf, kbuf, vbuf, rpb, relidx, mask, attn);
    gemm_proj_kernel<<<dim3(4, 784), 256, 0, stream>>>(attn, wproj_bf, b_proj, (float*)d_out);
}
